// Model_63926293234212
// MI455X (gfx1250) — matrix operand order repaired and checked by reading, not compiled, not run on hardware
//
#include <hip/hip_runtime.h>
#include <stdint.h>

typedef __attribute__((ext_vector_type(16))) _Float16 v16h;
typedef __attribute__((ext_vector_type(8)))  _Float16 v8h;
typedef __attribute__((ext_vector_type(4)))  _Float16 v4h;
typedef __attribute__((ext_vector_type(2)))  _Float16 v2h;
typedef __attribute__((ext_vector_type(8)))  float    v8f;
typedef __attribute__((ext_vector_type(4)))  float    v4f;
typedef __attribute__((ext_vector_type(4)))  unsigned int u32x4;
typedef __attribute__((ext_vector_type(8)))  int      i32x8;
typedef __attribute__((ext_vector_type(4)))  int      i32x4;

#define S_LEN   2048
#define D_DIM   64
#define KV_TILE 64
#define WAVES   4
#define SCALE   0.125f               // 1/sqrt(64), folded into Q fragments
#define KEEP_SCALE (1.0f/0.9f)       // folded into final 1/l
#define DROP_BYTE 26u                // 26/256 ~= 0.1016 drop prob

__device__ __forceinline__ int swz32(int x) {      // A-layout half-pattern swizzle
    return (x & ~24) | ((x & 8) << 1) | ((x & 16) >> 1);
}
__device__ __forceinline__ uint32_t hash4(uint32_t x) {
    x *= 0x9E3779B9u; x ^= x >> 16; x *= 0x85EBCA6Bu; x ^= x >> 13;
    return x;                                       // 4 Bernoulli bytes per hash
}
__device__ __forceinline__ uint32_t lds_off(const void* p) {
    return (uint32_t)(uintptr_t)p;                  // LDS aperture: offset in addr[31:0]
}
// packed f32->f16 convert (v_cvt_pk_rtz_f16_f32), re-typed to _Float16 vectors
__device__ __forceinline__ v2h pk2(float a, float b) {
    return __builtin_bit_cast(v2h, __builtin_amdgcn_cvt_pkrtz(a, b));
}
__device__ __forceinline__ v4h  cat2(v2h a, v2h b){ return __builtin_shufflevector(a,b,0,1,2,3); }
__device__ __forceinline__ v8h  cat4(v4h a, v4h b){ return __builtin_shufflevector(a,b,0,1,2,3,4,5,6,7); }
__device__ __forceinline__ v16h cat8(v8h a, v8h b){
    return __builtin_shufflevector(a,b,0,1,2,3,4,5,6,7,8,9,10,11,12,13,14,15);
}

// One TDM descriptor: 2D tensor (dim0=64 f32 contiguous, stride 64; dim1=2048 rows),
// tile 64x64 -> linear row-major into LDS at lds_addr. Wave-level op, EXEC ignored.
__device__ __forceinline__ void tdm_load_tile(uint32_t lds_addr, const void* gaddr) {
    uint64_t ga = (uint64_t)(uintptr_t)gaddr;
    u32x4 g0;
    g0[0] = 1u;                                          // count=1 (valid descriptor)
    g0[1] = lds_addr;                                    // lds_addr (bytes)
    g0[2] = (uint32_t)ga;                                // global_addr[31:0]
    g0[3] = (uint32_t)((ga >> 32) & 0x1FFFFFFu)          // global_addr[56:32]
          | 0x80000000u;                                 // type=2 ("image")
    i32x8 g1;
    g1[0] = 0x00020000;                                  // data_size=2 (4 bytes)
    g1[1] = (int)(64u << 16);                            // tensor_dim0[15:0]=64 @ bits63:48
    g1[2] = (int)(2048u << 16);                          // tensor_dim1[15:0]=2048 @ bits95:80
    g1[3] = (int)(64u << 16);                            // tile_dim0=64 @ bits127:112
    g1[4] = 64;                                          // tile_dim1=64 (tile_dim2=0)
    g1[5] = 64;                                          // tensor_dim0_stride=64 (lo32)
    g1[6] = 0;
    g1[7] = 0;
    i32x4 z4 = (i32x4){0,0,0,0};
    i32x8 z8 = (i32x8){0,0,0,0,0,0,0,0};
    __builtin_amdgcn_tensor_load_to_lds(g0, g1, z4, z4, z8, 0);
}

__global__ __launch_bounds__(128)
void flash_attn_wmma(const float* __restrict__ Q, const float* __restrict__ K,
                     const float* __restrict__ V, const float* __restrict__ Msk,
                     float* __restrict__ Out) {
    __shared__ alignas(16) float    Ksta[KV_TILE * D_DIM];      // f32 staging (TDM); O epilogue reuse
    __shared__ alignas(16) float    Vsta[KV_TILE * D_DIM];      // f32 staging (TDM)
    __shared__ alignas(32) _Float16 Kt[KV_TILE * D_DIM];        // [kv][d] f16
    __shared__ alignas(32) _Float16 Vt[D_DIM * KV_TILE];        // [d][kv] f16
    __shared__ alignas(32) _Float16 Pt[WAVES * 16 * KV_TILE];   // per-wave P^T [q][kv] f16

    const int wave = threadIdx.x >> 5;
    const int lane = threadIdx.x & 31;
    const int hi   = lane >> 4;
    const int ln   = lane & 15;

    const int q0 = blockIdx.x * (16 * WAVES) + wave * 16;
    const int bh = blockIdx.y;

    const float* Qb = Q + (size_t)bh * S_LEN * D_DIM;
    const float* Kb = K + (size_t)bh * S_LEN * D_DIM;
    const float* Vb = V + (size_t)bh * S_LEN * D_DIM;
    float*       Ob = Out + (size_t)bh * S_LEN * D_DIM;

    const uint32_t ksta0 = lds_off(&Ksta[0]);
    const uint32_t vsta0 = lds_off(&Vsta[0]);

    auto issue_tile = [&](int kt) {                  // wave0 drives the TDM
        if (wave == 0) {
            tdm_load_tile(ksta0, Kb + (size_t)kt * D_DIM);
            tdm_load_tile(vsta0, Vb + (size_t)kt * D_DIM);
        }
    };

    // ---- Q^T B-fragments (two 32x16 blocks), pre-scaled by 1/sqrt(d) ----
    v16h qb[2];
    #pragma unroll
    for (int j = 0; j < 2; ++j) {
        const float* qp = Qb + (size_t)(q0 + ln) * D_DIM + j * 32 + hi * 16;
        v2h p0 = pk2(qp[0]*SCALE,  qp[1]*SCALE);
        v2h p1 = pk2(qp[2]*SCALE,  qp[3]*SCALE);
        v2h p2 = pk2(qp[4]*SCALE,  qp[5]*SCALE);
        v2h p3 = pk2(qp[6]*SCALE,  qp[7]*SCALE);
        v2h p4 = pk2(qp[8]*SCALE,  qp[9]*SCALE);
        v2h p5 = pk2(qp[10]*SCALE, qp[11]*SCALE);
        v2h p6 = pk2(qp[12]*SCALE, qp[13]*SCALE);
        v2h p7 = pk2(qp[14]*SCALE, qp[15]*SCALE);
        qb[j] = cat8(cat4(cat2(p0,p1), cat2(p2,p3)), cat4(cat2(p4,p5), cat2(p6,p7)));
    }

    float m_run = -INFINITY, l_run = 0.0f;           // per-lane scalar stats (q = q0+ln)
    v8f oacc[4];
    #pragma unroll
    for (int n = 0; n < 4; ++n) oacc[n] = (v8f){};

    _Float16* Ptw = Pt + wave * 16 * KV_TILE;

    issue_tile(0);

    for (int kt = 0; kt < S_LEN; kt += KV_TILE) {
        if (wave == 0) __builtin_amdgcn_s_wait_tensorcnt(0);
        __syncthreads();

        // ---- convert staging f32 -> Kt [kv][d] / Vt [d][kv] ----
        #pragma unroll
        for (int j = 0; j < 8; ++j) {
            int i   = threadIdx.x + j * 128;
            int row = i >> 4;                         // kv
            int c4  = (i & 15) << 2;                  // d
            v4f kf = *(const v4f*)(&Ksta[row * D_DIM + c4]);
            v4h kh = cat2(pk2(kf[0], kf[1]), pk2(kf[2], kf[3]));
            *(v4h*)(&Kt[row * D_DIM + c4]) = kh;
            v4f vf = *(const v4f*)(&Vsta[row * D_DIM + c4]);
            int srow = row;
            Vt[(c4 + 0) * KV_TILE + srow] = (_Float16)vf[0];
            Vt[(c4 + 1) * KV_TILE + srow] = (_Float16)vf[1];
            Vt[(c4 + 2) * KV_TILE + srow] = (_Float16)vf[2];
            Vt[(c4 + 3) * KV_TILE + srow] = (_Float16)vf[3];
        }
        __syncthreads();

        if (kt + KV_TILE < S_LEN) issue_tile(kt + KV_TILE);   // prefetch next tile

        // ---- S^T = K * (Q*scale)^T + mask^T (mask as WMMA C-init) ----
        float sc[4][8];
        #pragma unroll
        for (int nb = 0; nb < 4; ++nb) {
            const v16h ak0 = *(const v16h*)(&Kt[(nb * 16 + ln) * D_DIM + hi * 16]);
            const v16h ak1 = *(const v16h*)(&Kt[(nb * 16 + ln) * D_DIM + 32 + hi * 16]);
            const float* mp = Msk + (size_t)(q0 + ln) * S_LEN + kt + nb * 16 + hi * 8;
            v4f m0 = *(const v4f*)(mp);
            v4f m1 = *(const v4f*)(mp + 4);
            v8f c;
            c[0]=m0[0]; c[1]=m0[1]; c[2]=m0[2]; c[3]=m0[3];
            c[4]=m1[0]; c[5]=m1[1]; c[6]=m1[2]; c[7]=m1[3];
            c = __builtin_amdgcn_wmma_f32_16x16x32_f16(false, ak0, false, qb[0],
                                                       (short)0, c, false, false);
            c = __builtin_amdgcn_wmma_f32_16x16x32_f16(false, ak1, false, qb[1],
                                                       (short)0, c, false, false);
            #pragma unroll
            for (int r = 0; r < 8; ++r) sc[nb][r] = c[r];
        }

        // ---- online softmax (per-lane scalar stats) ----
        float t = sc[0][0];
        #pragma unroll
        for (int nb = 0; nb < 4; ++nb)
            #pragma unroll
            for (int r = 0; r < 8; ++r) t = fmaxf(t, sc[nb][r]);
        t = fmaxf(t, __shfl_xor(t, 16, 32));          // combine kv lane-halves

        if (__any(t > m_run)) {                        // rescale only when max moved
            float mnew = fmaxf(m_run, t);
            float fs = __expf(m_run - mnew);
            m_run = mnew;
            l_run *= fs;
            #pragma unroll
            for (int n = 0; n < 4; ++n)
                #pragma unroll
                for (int r = 0; r < 8; ++r) oacc[n][r] *= fs;
        }

        float su = 0.0f;
        #pragma unroll
        for (int nb = 0; nb < 4; ++nb)
            #pragma unroll
            for (int r = 0; r < 8; ++r) {
                float p = __expf(sc[nb][r] - m_run);
                sc[nb][r] = p;
                su += p;
            }
        su += __shfl_xor(su, 16, 32);
        l_run += su;                                   // denominator: un-dropped sums

        // ---- dropout (1 hash -> 4 decisions) + stage P^T: 4 x ds_store_b128 ----
        #pragma unroll
        for (int nb = 0; nb < 4; ++nb) {
            float pv[8];
            #pragma unroll
            for (int rg = 0; rg < 2; ++rg) {
                uint32_t kvb = (uint32_t)(kt + nb * 16 + hi * 8 + rg * 4);
                uint32_t idx = ((uint32_t)bh * S_LEN + (uint32_t)(q0 + ln)) * S_LEN + kvb;
                uint32_t h = hash4(idx);
                #pragma unroll
                for (int rr = 0; rr < 4; ++rr) {
                    int r = rg * 4 + rr;
                    bool drop = ((h >> (rr * 8)) & 0xFFu) < DROP_BYTE;
                    pv[r] = drop ? 0.0f : sc[nb][r];
                }
            }
            v8h ph = cat4(cat2(pk2(pv[0], pv[1]), pk2(pv[2], pv[3])),
                          cat2(pk2(pv[4], pv[5]), pk2(pv[6], pv[7])));
            *(v8h*)(&Ptw[ln * KV_TILE + nb * 16 + hi * 8]) = ph;
        }
        asm volatile("s_wait_dscnt 0" ::: "memory");   // wave-local LDS RAW

        // ---- O^T += V^T (A-frags, plain kv) * P^T (B-frags, plain kv) ----
        #pragma unroll
        for (int kb = 0; kb < 2; ++kb) {
            const v16h pb = *(const v16h*)(&Ptw[ln * KV_TILE + kb * 32 + hi * 16]);
            #pragma unroll
            for (int nd = 0; nd < 4; ++nd) {
                const v16h av = *(const v16h*)(&Vt[(nd * 16 + ln) * KV_TILE
                                                   + kb * 32 + hi * 16]);
                oacc[nd] = __builtin_amdgcn_wmma_f32_16x16x32_f16(false, av, false, pb,
                                                                  (short)0, oacc[nd],
                                                                  false, false);
            }
        }
    }

    // ---- epilogue: O^T -> O via LDS (reuse staging), coalesced b128 stores ----
    float invl = KEEP_SCALE / l_run;
    float* Olds = Ksta + wave * (16 * D_DIM);
    #pragma unroll
    for (int nd = 0; nd < 4; ++nd)
        #pragma unroll
        for (int g = 0; g < 2; ++g) {
            v4f tv;
            #pragma unroll
            for (int i = 0; i < 4; ++i) tv[i] = oacc[nd][g * 4 + i] * invl;
            *(v4f*)(&Olds[ln * D_DIM + nd * 16 + hi * 8 + g * 4]) = tv;
        }
    asm volatile("s_wait_dscnt 0" ::: "memory");
    #pragma unroll
    for (int j = 0; j < 4; ++j) {
        int fid = lane + j * 32;
        int row = fid >> 4;
        int c4  = (fid & 15) << 2;
        v4f tv = *(const v4f*)(&Olds[row * D_DIM + c4]);
        *(v4f*)(&Ob[(size_t)(q0 + row) * D_DIM + c4]) = tv;
    }
}

extern "C" void kernel_launch(void* const* d_in, const int* in_sizes, int n_in,
                              void* d_out, int out_size, void* d_ws, size_t ws_size,
                              hipStream_t stream) {
    const float* Q   = (const float*)d_in[0];
    const float* K   = (const float*)d_in[1];
    const float* V   = (const float*)d_in[2];
    const float* Msk = (const float*)d_in[3];
    float* Out = (float*)d_out;

    int bh_count = in_sizes[0] / (S_LEN * D_DIM);   // B*H = 64
    dim3 grid(S_LEN / (16 * WAVES), bh_count);      // (32, 64)
    dim3 block(128);                                // 4 waves (wave32)
    hipLaunchKernelGGL(flash_attn_wmma, grid, block, 0, stream, Q, K, V, Msk, Out);
}
